// PREG_76605036692263
// MI455X (gfx1250) — matrix operand order repaired and checked by reading, not compiled, not run on hardware
//
#include <hip/hip_runtime.h>

// The Python harness owns main(), I/O, device malloc/memcpy, and timing.

typedef __attribute__((ext_vector_type(16))) _Float16 v16h;
typedef __attribute__((ext_vector_type(8)))  _Float16 v8h;
typedef __attribute__((ext_vector_type(8)))  float    v8f;
typedef __attribute__((ext_vector_type(4)))  float    v4f;

namespace {
constexpr int   kH = 64, kN = 1024, kD = 128;
constexpr int   kSplit        = 8;              // workgroups per head
constexpr int   kBlocks       = kH * kSplit;    // 512
constexpr int   kThreads      = 256;            // 8 wave32s per workgroup
constexpr int   kRowsPerBlock = kN / kSplit;    // 128 rows -> one 16-row tile per wave
constexpr int   kXPitch       = 1032;           // halfs per LDS x^T row (pad 1024 -> bank spread)
constexpr float kThrSq        = 0.09f;          // (0.3)^2 for squared center distance
constexpr float kThr          = 0.3f;
constexpr float kScale        = 1.0f / (1024.0f * 128.0f * 1024.0f); // /(N*D)/N per head

// LDS layout (bytes)
constexpr int kSmemXT    = kD * kXPitch * 2;    // 264192: x^T as f16, [D][kXPitch]
constexpr int kSmemCX    = kSmemXT;             // cx[1024] f32
constexpr int kSmemCY    = kSmemCX + kN * 4;    // cy[1024] f32
constexpr int kSmemY     = kSmemCY + kN * 4;    // y [1024] f32
constexpr int kSmemWsum  = kSmemY + kN * 4;     // per-wave partial sums
constexpr int kSmemTotal = kSmemWsum + 8 * 4;   // 276512 < 320KB WGP LDS
static_assert(kSmemTotal <= 320 * 1024, "LDS budget");
} // namespace

__global__ __launch_bounds__(kThreads, 1)
void gcn_head_loss(const float* __restrict__ x,
                   const float* __restrict__ coords,
                   const float* __restrict__ y,
                   float* __restrict__ partial) {
  extern __shared__ char smem[];
  _Float16* sxT   = reinterpret_cast<_Float16*>(smem);
  float*    scx   = reinterpret_cast<float*>(smem + kSmemCX);
  float*    scy   = reinterpret_cast<float*>(smem + kSmemCY);
  float*    sy    = reinterpret_cast<float*>(smem + kSmemY);
  float*    swsum = reinterpret_cast<float*>(smem + kSmemWsum);

  const int tid      = threadIdx.x;
  const int head     = blockIdx.x / kSplit;
  const int rowBase0 = (blockIdx.x % kSplit) * kRowsPerBlock;

  const float* xh = x      + (size_t)head * kN * kD;
  const float* ch = coords + (size_t)head * kN * 4;
  const float* yh = y      + (size_t)head * kN;

  // ---- Stage centers + y into LDS ----
  for (int n = tid; n < kN; n += kThreads) {
    v4f c = *(const v4f*)&ch[n * 4];
    scx[n] = (c.x + c.z) * 0.5f;
    scy[n] = (c.y + c.w) * 0.5f;
    sy[n]  = yh[n];
  }
  // ---- Stage x^T (f32 -> f16, transposed with padded pitch) ----
  {
    const int dq = (tid & 31) * 4;            // 32 lanes cover D=128 as float4
    for (int k = (tid >> 5); k < kN; k += 8) {
      v4f xv = *(const v4f*)&xh[k * kD + dq];
      sxT[(dq + 0) * kXPitch + k] = (_Float16)xv.x;
      sxT[(dq + 1) * kXPitch + k] = (_Float16)xv.y;
      sxT[(dq + 2) * kXPitch + k] = (_Float16)xv.z;
      sxT[(dq + 3) * kXPitch + k] = (_Float16)xv.w;
    }
  }
  __syncthreads();

  const int wave = tid >> 5;
  const int lane = tid & 31;
  const int g    = lane >> 4;   // lane half: selects A K-subset / B K-half / C row-half
  const int m    = lane & 15;   // A row within tile; B/C column within tile
  const int rowBase = rowBase0 + wave * 16;
  const int rowI    = rowBase + m;

  const float xiCx = scx[rowI];
  const float xiCy = scy[rowI];
  const float xiY  = sy[rowI];

  v8f zero = {};
  v8f acc[8];
#pragma unroll
  for (int dt = 0; dt < 8; ++dt) acc[dt] = zero;
  int deg4 = 0;   // 4x the degree count (popcount of 0x3C00 nibble patterns)

  union F16u { v4f v[4]; float f[16]; };
  union Au   { v16h h; unsigned u[8]; };
  union Bu   { v16h h; v8h p[2]; };

#pragma unroll 2
  for (int kt = 0; kt < kN / 32; ++kt) {
    // A-fragment K coordinates for this lane: j = j0 + e (e<8), j0 + 16 + (e-8) (e>=8)
    const int j0 = kt * 32 + 8 * g;
    F16u CX, CY, YV;
    {
      const v4f* p = (const v4f*)&scx[j0];
      CX.v[0] = p[0]; CX.v[1] = p[1]; CX.v[2] = p[4]; CX.v[3] = p[5];
    }
    {
      const v4f* p = (const v4f*)&scy[j0];
      CY.v[0] = p[0]; CY.v[1] = p[1]; CY.v[2] = p[4]; CY.v[3] = p[5];
    }
    {
      const v4f* p = (const v4f*)&sy[j0];
      YV.v[0] = p[0]; YV.v[1] = p[1]; YV.v[2] = p[4]; YV.v[3] = p[5];
    }

    // Build adjacency fragment in registers: {0,1} exact in f16.
    Au A;
#pragma unroll
    for (int ee = 0; ee < 8; ++ee) {
      const int e0 = 2 * ee, e1 = 2 * ee + 1;
      float dx0 = xiCx - CX.f[e0], dy0 = xiCy - CY.f[e0], dv0 = xiY - YV.f[e0];
      float dx1 = xiCx - CX.f[e1], dy1 = xiCy - CY.f[e1], dv1 = xiY - YV.f[e1];
      bool b0 = (dx0 * dx0 + dy0 * dy0 < kThrSq) && (__builtin_fabsf(dv0) < kThr);
      bool b1 = (dx1 * dx1 + dy1 * dy1 < kThrSq) && (__builtin_fabsf(dv1) < kThr);
      unsigned w = (b0 ? 0x3C00u : 0u) | (b1 ? 0x3C000000u : 0u);
      A.u[ee] = w;
      deg4 += __builtin_popcount(w);   // v_bcnt: 4 bits per set element
    }

    // One A fragment feeds 8 WMMAs (all D tiles). B: two ds_load_b128 at k 8g and 16+8g.
    const int kB = kt * 32 + 8 * g;
#pragma unroll
    for (int dt = 0; dt < 8; ++dt) {
      const int d = dt * 16 + m;
      Bu B;
      const v8h* pb = (const v8h*)&sxT[d * kXPitch + kB];
      B.p[0] = pb[0];
      B.p[1] = pb[2];
      acc[dt] = __builtin_amdgcn_wmma_f32_16x16x32_f16(
          false, A.h, false, B.h, (short)0, acc[dt], false, false);
    }
  }

  // Degree of row m = count(lane m) + count(lane m+16)  (the two K-subsets).
  float pdeg   = (float)(deg4 >> 2);
  float degRow = pdeg + __shfl_xor(pdeg, 16, 32);   // lane L -> deg[L & 15]
  float invDeg[8];
#pragma unroll
  for (int r = 0; r < 8; ++r) {
    // deg is a small positive integer (>=1, diagonal always set); v_rcp_f32 is plenty.
    invDeg[r] = __builtin_amdgcn_rcpf(__shfl(degRow, 8 * g + r, 32));
  }

  // diff = (A@x)/deg - x ; accumulate squared error.
  float lacc = 0.0f;
#pragma unroll
  for (int dt = 0; dt < 8; ++dt) {
#pragma unroll
    for (int r = 0; r < 8; ++r) {
      const int row  = rowBase + 8 * g + r;         // C layout: VGPR r, lane-half g
      const float xv = xh[row * kD + dt * 16 + m];
      const float df = __builtin_fmaf(acc[dt][r], invDeg[r], -xv);
      lacc += df * df;
    }
  }

  // Deterministic in-block reduction: wave shuffle tree, then wave 0 sums.
#pragma unroll
  for (int off = 16; off > 0; off >>= 1) lacc += __shfl_xor(lacc, off, 32);
  if (lane == 0) swsum[wave] = lacc;
  __syncthreads();
  if (tid == 0) {
    float t = 0.0f;
#pragma unroll
    for (int w = 0; w < 8; ++w) t += swsum[w];
    partial[blockIdx.x] = t;
  }
}

__global__ __launch_bounds__(kThreads, 1)
void reduce_partials(const float* __restrict__ partial, float* __restrict__ out) {
  __shared__ float s[kThreads];
  const int tid = threadIdx.x;
  s[tid] = partial[tid] + partial[tid + kThreads];   // 512 partials, fixed order
  __syncthreads();
  for (int off = kThreads / 2; off > 0; off >>= 1) {
    if (tid < off) s[tid] += s[tid + off];
    __syncthreads();
  }
  if (tid == 0) out[0] = s[0] * kScale;
}

extern "C" void kernel_launch(void* const* d_in, const int* in_sizes, int n_in,
                              void* d_out, int out_size, void* d_ws, size_t ws_size,
                              hipStream_t stream) {
  (void)in_sizes; (void)n_in; (void)out_size; (void)ws_size;
  const float* x  = (const float*)d_in[0];   // [H*N, D] f32
  const float* hc = (const float*)d_in[1];   // [H, N, 4] f32
  const float* yt = (const float*)d_in[2];   // [H*N] f32
  float* out      = (float*)d_out;           // scalar f32
  float* partial  = (float*)d_ws;            // kBlocks floats of scratch

  // 276.5 KB dynamic LDS per workgroup (fits CDNA5's 320KB WGP LDS).
  hipFuncSetAttribute(reinterpret_cast<const void*>(gcn_head_loss),
                      hipFuncAttributeMaxDynamicSharedMemorySize, kSmemTotal);

  gcn_head_loss<<<kBlocks, kThreads, kSmemTotal, stream>>>(x, hc, yt, partial);
  reduce_partials<<<1, kThreads, 0, stream>>>(partial, out);
}
